// SwinTransformerBlock_29738353557858
// MI455X (gfx1250) — compile-verified, matrix operand order checked by reading, not run on hardware
//
#include <hip/hip_runtime.h>
#include <hip/hip_bf16.h>

// ---------------------------------------------------------------------------
// Swin Transformer block, MI455X (gfx1250), wave32 + WMMA bf16.
// B=32 H=56 W=56 C=512 HEADS=16 WS=7 SS=3 N=49 HD=32 HID=2048 nW=64
// ---------------------------------------------------------------------------

#define Bc   32
#define Hc   56
#define Wc   56
#define Cc   512
#define HEADS 16
#define WSc  7
#define SSc  3
#define Nc   49
#define HDc  32
#define HIDc 2048
#define nWh  8
#define nWw  8
#define nWc  64
#define ROWS (Bc * nWc * Nc)      // 100352 token rows

typedef __attribute__((ext_vector_type(16))) __bf16 bfx16;
typedef __attribute__((ext_vector_type(8)))  float  fx8;

union FragB {
    bfx16 v;
    uint4 q[2];
};

__device__ __forceinline__ unsigned short f2bf(float f) {
    unsigned u = __float_as_uint(f);
    unsigned r = u + 0x7FFFu + ((u >> 16) & 1u);
    return (unsigned short)(r >> 16);
}
__device__ __forceinline__ float bf2f(unsigned short h) {
    return __uint_as_float(((unsigned)h) << 16);
}

// ---------------------------------------------------------------------------
// fp32 -> bf16 conversion (weights)
// ---------------------------------------------------------------------------
__global__ void cvt_bf16_kernel(const float* __restrict__ in,
                                unsigned short* __restrict__ out, int n) {
    int i = blockIdx.x * blockDim.x + threadIdx.x;
    if (i < n) out[i] = f2bf(in[i]);
}

// ---------------------------------------------------------------------------
// LayerNorm (+ optional cyclic-shift + window partition gather) -> bf16 rows.
// one 512-wide row per 256-thread block (8 waves).
// ---------------------------------------------------------------------------
__global__ __launch_bounds__(256)
void ln_kernel(const float* __restrict__ in, const float* __restrict__ gam,
               const float* __restrict__ bet, unsigned short* __restrict__ out,
               int windowed) {
    __shared__ float rs[256];
    __shared__ float rq[256];
    const int tid = threadIdx.x;
    const int row = blockIdx.x;
    long src;
    if (windowed) {
        int b = row / (nWc * Nc);
        int rem = row % (nWc * Nc);
        int widx = rem / Nc, n = rem % Nc;
        int wh = widx / nWw, ww = widx % nWw;
        int iy = n / WSc, ix = n % WSc;
        int h = (wh * WSc + iy + SSc) % Hc;    // roll(-SS) gather
        int w = (ww * WSc + ix + SSc) % Wc;
        src = ((long)(b * Hc + h) * Wc + w) * Cc;
    } else {
        src = (long)row * Cc;
    }
    float v0 = in[src + tid];
    float v1 = in[src + tid + 256];
    rs[tid] = v0 + v1;
    rq[tid] = v0 * v0 + v1 * v1;
    __syncthreads();
    for (int off = 128; off > 0; off >>= 1) {
        if (tid < off) { rs[tid] += rs[tid + off]; rq[tid] += rq[tid + off]; }
        __syncthreads();
    }
    float mean = rs[0] * (1.0f / Cc);
    float var  = rq[0] * (1.0f / Cc) - mean * mean;
    float rstd = rsqrtf(var + 1e-5f);
    long dst = (long)row * Cc;
    out[dst + tid]       = f2bf((v0 - mean) * rstd * gam[tid] + bet[tid]);
    out[dst + tid + 256] = f2bf((v1 - mean) * rstd * gam[tid + 256] + bet[tid + 256]);
}

// ---------------------------------------------------------------------------
// Generic TN GEMM:  C[M,N] = A[M,K](bf16, row-major) * W[N,K](bf16)^T + bias
// 128x64 block tile, 8 waves, each wave = 32x32 output via 4 accumulators
// (2 A-frags x 2 B-frags -> 4 WMMA per 8 ds_load_b128), LDS double-buffered
// (one barrier per 32-wide K-step).
// modes: 0 = bf16 out             (qkv)
//        1 = bf16 out, GELU       (fc1)
//        2 = f32 out, + res[row]  (fc2 -> d_out)
//        3 = f32 out, window-reverse scatter + shortcut (proj -> x_res)
// ---------------------------------------------------------------------------
__global__ __launch_bounds__(256)
void gemm_bf16_kernel(const unsigned short* __restrict__ A,
                      const unsigned short* __restrict__ Wt,
                      const float* __restrict__ bias,
                      int M, int N, int K, int mode,
                      unsigned short* __restrict__ outh,
                      float* __restrict__ outf,
                      const float* __restrict__ res) {
    __shared__ __align__(16) unsigned short As[2][128 * 40];
    __shared__ __align__(16) unsigned short Bs[2][64 * 40];

    const int tid  = threadIdx.x;
    const int lane = tid & 31;
    const int wave = tid >> 5;
    const int wm = wave & 3;       // 0..3 : 32-row M subtile
    const int wn = wave >> 2;      // 0..1 : 32-col N subtile
    const int m0 = blockIdx.y * 128;
    const int n0 = blockIdx.x * 64;
    const int srow = tid >> 2;            // staging row (A: srow & srow+64)
    const int scol = (tid & 3) * 8;       // staging col chunk

    const int kb    = (lane >= 16) ? 8 : 0;
    const int arow0 = (wm * 32 + (lane & 15)) * 40;
    const int arow1 = (wm * 32 + 16 + (lane & 15)) * 40;
    const int brow0 = (wn * 32 + (lane & 15)) * 40;
    const int brow1 = (wn * 32 + 16 + (lane & 15)) * 40;

    fx8 acc00 = {}, acc01 = {}, acc10 = {}, acc11 = {};

    // prologue: stage K-slice 0 into buffer 0
    {
        const uint4 a0 = *(const uint4*)(A  + (long)(m0 + srow) * K + scol);
        const uint4 a1 = *(const uint4*)(A  + (long)(m0 + srow + 64) * K + scol);
        const uint4 b0 = *(const uint4*)(Wt + (long)(n0 + srow) * K + scol);
        *(uint4*)(As[0] + srow * 40 + scol)        = a0;
        *(uint4*)(As[0] + (srow + 64) * 40 + scol) = a1;
        *(uint4*)(Bs[0] + srow * 40 + scol)        = b0;
    }

    int cur = 0;
    for (int kt = 0; kt < K; kt += 32) {
        __syncthreads();   // buffer `cur` ready; buffer cur^1 free to overwrite

        const bool hasNext = (kt + 32) < K;
        uint4 a0n = {}, a1n = {}, bn = {};
        if (hasNext) {
            a0n = *(const uint4*)(A  + (long)(m0 + srow) * K + kt + 32 + scol);
            a1n = *(const uint4*)(A  + (long)(m0 + srow + 64) * K + kt + 32 + scol);
            bn  = *(const uint4*)(Wt + (long)(n0 + srow) * K + kt + 32 + scol);
            if (kt + 64 < K) {
                __builtin_prefetch(A  + (long)(m0 + srow) * K + kt + 64 + scol, 0, 1);
                __builtin_prefetch(Wt + (long)(n0 + srow) * K + kt + 64 + scol, 0, 1);
            }
        }

        FragB a0, a1, b0, b1;
        const unsigned short* as = As[cur];
        const unsigned short* bs = Bs[cur];
        a0.q[0] = *(const uint4*)(as + arow0 + kb);
        a0.q[1] = *(const uint4*)(as + arow0 + kb + 16);
        a1.q[0] = *(const uint4*)(as + arow1 + kb);
        a1.q[1] = *(const uint4*)(as + arow1 + kb + 16);
        b0.q[0] = *(const uint4*)(bs + brow0 + kb);
        b0.q[1] = *(const uint4*)(bs + brow0 + kb + 16);
        b1.q[0] = *(const uint4*)(bs + brow1 + kb);
        b1.q[1] = *(const uint4*)(bs + brow1 + kb + 16);

        acc00 = __builtin_amdgcn_wmma_f32_16x16x32_bf16(false, a0.v, false, b0.v,
                                                        (short)0, acc00, false, false);
        acc01 = __builtin_amdgcn_wmma_f32_16x16x32_bf16(false, a0.v, false, b1.v,
                                                        (short)0, acc01, false, false);
        acc10 = __builtin_amdgcn_wmma_f32_16x16x32_bf16(false, a1.v, false, b0.v,
                                                        (short)0, acc10, false, false);
        acc11 = __builtin_amdgcn_wmma_f32_16x16x32_bf16(false, a1.v, false, b1.v,
                                                        (short)0, acc11, false, false);

        if (hasNext) {
            unsigned short* asn = As[cur ^ 1];
            unsigned short* bsn = Bs[cur ^ 1];
            *(uint4*)(asn + srow * 40 + scol)        = a0n;
            *(uint4*)(asn + (srow + 64) * 40 + scol) = a1n;
            *(uint4*)(bsn + srow * 40 + scol)        = bn;
        }
        cur ^= 1;
    }

    // epilogue: 4 accumulators cover rows wm*32..+31, cols wn*32..+31
    for (int ti = 0; ti < 2; ++ti) {
        for (int tj = 0; tj < 2; ++tj) {
            const fx8 acc = ti ? (tj ? acc11 : acc10) : (tj ? acc01 : acc00);
            const int rbase = m0 + wm * 32 + ti * 16 + ((lane >= 16) ? 8 : 0);
            const int col   = n0 + wn * 32 + tj * 16 + (lane & 15);
            const float bc = bias[col];
            for (int i = 0; i < 8; ++i) {
                const int row = rbase + i;
                float v = acc[i] + bc;
                if (mode == 0) {
                    outh[(long)row * N + col] = f2bf(v);
                } else if (mode == 1) {
                    float g = 0.5f * v * (1.0f + erff(v * 0.70710678118f));
                    outh[(long)row * N + col] = f2bf(g);
                } else if (mode == 2) {
                    outf[(long)row * N + col] = v + res[(long)row * N + col];
                } else {  // mode 3: window reverse + roll(+SS) + shortcut
                    int b = row / (nWc * Nc);
                    int rem = row % (nWc * Nc);
                    int widx = rem / Nc, n = rem % Nc;
                    int wh = widx / nWw, ww = widx % nWw;
                    int iy = n / WSc, ix = n % WSc;
                    int h = (wh * WSc + iy + SSc) % Hc;
                    int w = (ww * WSc + ix + SSc) % Wc;
                    long p = (long)(b * Hc + h) * Wc + w;
                    outf[p * Cc + col] = v + res[p * Cc + col];
                }
            }
        }
    }
}

// ---------------------------------------------------------------------------
// Windowed attention: one block per (window, head). N=49 padded to 64.
// scores and attn@v both via v_wmma_f32_16x16x32_bf16. 128 threads (4 waves).
// ---------------------------------------------------------------------------
__global__ __launch_bounds__(128)
void attn_kernel(const unsigned short* __restrict__ qkv,
                 const float* __restrict__ relt,
                 unsigned short* __restrict__ o) {
    __shared__ __align__(16) unsigned short Qs[64 * 40];
    __shared__ __align__(16) unsigned short Ks[64 * 40];
    __shared__ __align__(16) unsigned short Vt[32 * 72];   // transposed V: [d][tok]
    __shared__ float Ss[64 * 65];
    __shared__ __align__(16) unsigned short Sb[64 * 72];
    __shared__ float biasS[169];
    __shared__ int regS[64];

    const int tid  = threadIdx.x;
    const int lane = tid & 31;
    const int wave = tid >> 5;       // 0..3
    const int win  = blockIdx.x >> 4;
    const int head = blockIdx.x & 15;
    const int widx = win & (nWc - 1);
    const long qbase = (long)win * Nc * (3 * Cc) + head * HDc;

    // stage q,k,v (zero-pad tokens 49..63)
    for (int e = tid; e < 64 * 32; e += 128) {
        int tok = e >> 5, d = e & 31;
        unsigned short qv = 0, kv = 0, vv = 0;
        if (tok < Nc) {
            long p = qbase + (long)tok * (3 * Cc) + d;
            qv = qkv[p];
            kv = qkv[p + Cc];
            vv = qkv[p + 2 * Cc];
        }
        Qs[tok * 40 + d] = qv;
        Ks[tok * 40 + d] = kv;
        Vt[d * 72 + tok] = vv;
    }
    // stage per-head relative-position bias column
    for (int t = tid; t < 169; t += 128) biasS[t] = relt[t * HEADS + head];
    // shifted-window region ids (mask)
    if (tid < 64) {
        int r = 0;
        if (tid < Nc) {
            int iy = tid / WSc, ix = tid % WSc;
            int h = (widx / nWw) * WSc + iy;
            int w = (widx % nWw) * WSc + ix;
            int rh = h < (Hc - WSc) ? 0 : (h < (Hc - SSc) ? 1 : 2);
            int rw = w < (Wc - WSc) ? 0 : (w < (Wc - SSc) ? 1 : 2);
            r = rh * 3 + rw;
        }
        regS[tid] = r;
    }
    __syncthreads();

    const int kb = (lane >= 16) ? 8 : 0;

    // scores: S = (Q * K^T) * scale + bias + mask   (each wave: 16 rows x 64 cols)
    {
        FragB a;
        const int arow = (wave * 16 + (lane & 15)) * 40;
        a.q[0] = *(const uint4*)(Qs + arow + kb);
        a.q[1] = *(const uint4*)(Qs + arow + kb + 16);
        for (int t = 0; t < 4; ++t) {
            FragB b;
            const int brow = (t * 16 + (lane & 15)) * 40;
            b.q[0] = *(const uint4*)(Ks + brow + kb);
            b.q[1] = *(const uint4*)(Ks + brow + kb + 16);
            fx8 acc = {};
            acc = __builtin_amdgcn_wmma_f32_16x16x32_bf16(false, a.v, false, b.v,
                                                          (short)0, acc, false, false);
            const int mb = wave * 16 + ((lane >= 16) ? 8 : 0);
            const int n  = t * 16 + (lane & 15);
            for (int i = 0; i < 8; ++i) {
                const int m = mb + i;
                float s;
                if (n >= Nc) {
                    s = -1e30f;
                } else if (m >= Nc) {
                    s = 0.0f;
                } else {
                    int ry = m / WSc - n / WSc + (WSc - 1);
                    int rx = m % WSc - n % WSc + (WSc - 1);
                    float msk = (regS[m] == regS[n]) ? 0.0f : -100.0f;
                    s = acc[i] * 0.17677669529f + biasS[ry * 13 + rx] + msk;
                }
                Ss[m * 65 + n] = s;
            }
        }
    }
    __syncthreads();

    // softmax, one row per lane; write bf16 probs (padded rows/cols -> 0)
    if (tid < 64) {
        const int r = tid;
        if (r < Nc) {
            float mx = -1e30f;
            for (int c = 0; c < Nc; ++c) mx = fmaxf(mx, Ss[r * 65 + c]);
            float sum = 0.0f;
            for (int c = 0; c < Nc; ++c) sum += __expf(Ss[r * 65 + c] - mx);
            float inv = 1.0f / sum;
            for (int c = 0; c < Nc; ++c)
                Sb[r * 72 + c] = f2bf(__expf(Ss[r * 65 + c] - mx) * inv);
            for (int c = Nc; c < 64; ++c) Sb[r * 72 + c] = 0;
        } else {
            for (int c = 0; c < 64; ++c) Sb[r * 72 + c] = 0;
        }
    }
    __syncthreads();

    // O = P * V   (each wave: 16 rows x 32 dims, K = 64 in two WMMA steps)
    {
        fx8 acc0 = {};
        fx8 acc1 = {};
        for (int kt = 0; kt < 2; ++kt) {
            FragB a;
            const int arow = (wave * 16 + (lane & 15)) * 72 + kt * 32;
            a.q[0] = *(const uint4*)(Sb + arow + kb);
            a.q[1] = *(const uint4*)(Sb + arow + kb + 16);
            FragB b0, b1;
            const int br0 = ((lane & 15)) * 72 + kt * 32;
            const int br1 = (16 + (lane & 15)) * 72 + kt * 32;
            b0.q[0] = *(const uint4*)(Vt + br0 + kb);
            b0.q[1] = *(const uint4*)(Vt + br0 + kb + 16);
            b1.q[0] = *(const uint4*)(Vt + br1 + kb);
            b1.q[1] = *(const uint4*)(Vt + br1 + kb + 16);
            acc0 = __builtin_amdgcn_wmma_f32_16x16x32_bf16(false, a.v, false, b0.v,
                                                           (short)0, acc0, false, false);
            acc1 = __builtin_amdgcn_wmma_f32_16x16x32_bf16(false, a.v, false, b1.v,
                                                           (short)0, acc1, false, false);
        }
        const int mb = wave * 16 + ((lane >= 16) ? 8 : 0);
        for (int t = 0; t < 2; ++t) {
            const fx8 acc = t ? acc1 : acc0;
            const int d = t * 16 + (lane & 15);
            for (int i = 0; i < 8; ++i) {
                const int m = mb + i;
                if (m < Nc)
                    o[((long)win * Nc + m) * Cc + head * HDc + d] = f2bf(acc[i]);
            }
        }
    }
}

// ---------------------------------------------------------------------------
// host-side launcher
// ---------------------------------------------------------------------------
extern "C" void kernel_launch(void* const* d_in, const int* in_sizes, int n_in,
                              void* d_out, int out_size, void* d_ws, size_t ws_size,
                              hipStream_t stream) {
    const float* x      = (const float*)d_in[0];
    const float* n1g    = (const float*)d_in[1];
    const float* n1b    = (const float*)d_in[2];
    const float* qkv_w  = (const float*)d_in[3];
    const float* qkv_b  = (const float*)d_in[4];
    const float* relt   = (const float*)d_in[5];
    const float* proj_w = (const float*)d_in[6];
    const float* proj_b = (const float*)d_in[7];
    const float* n2g    = (const float*)d_in[8];
    const float* n2b    = (const float*)d_in[9];
    const float* fc1_w  = (const float*)d_in[10];
    const float* fc1_b  = (const float*)d_in[11];
    const float* fc2_w  = (const float*)d_in[12];
    const float* fc2_b  = (const float*)d_in[13];

    char* ws = (char*)d_ws;
    // bf16 weight copies
    const size_t OFF_WQKV  = 0;                                  // 3C x C
    const size_t OFF_WPROJ = OFF_WQKV  + (size_t)3 * Cc * Cc * 2;
    const size_t OFF_WFC1  = OFF_WPROJ + (size_t)Cc * Cc * 2;
    const size_t OFF_WFC2  = OFF_WFC1  + (size_t)HIDc * Cc * 2;
    // R1: yw -> o -> h  (ROWS x C bf16)
    const size_t OFF_R1    = OFF_WFC2  + (size_t)Cc * HIDc * 2;
    // R2: qkv (ROWS x 3C bf16) -> x_res (ROWS x C f32)
    const size_t OFF_R2    = OFF_R1 + (size_t)ROWS * Cc * 2;
    // R3: fc1 activations (ROWS x HID bf16)
    const size_t OFF_R3    = OFF_R2 + (size_t)ROWS * 3 * Cc * 2;

    unsigned short* wqkv  = (unsigned short*)(ws + OFF_WQKV);
    unsigned short* wproj = (unsigned short*)(ws + OFF_WPROJ);
    unsigned short* wfc1  = (unsigned short*)(ws + OFF_WFC1);
    unsigned short* wfc2  = (unsigned short*)(ws + OFF_WFC2);
    unsigned short* yw    = (unsigned short*)(ws + OFF_R1);   // LN1 windows
    unsigned short* obuf  = (unsigned short*)(ws + OFF_R1);   // attn output (aliases yw)
    unsigned short* hbuf  = (unsigned short*)(ws + OFF_R1);   // LN2 output (aliases o)
    unsigned short* qkvb  = (unsigned short*)(ws + OFF_R2);
    float*          x_res = (float*)(ws + OFF_R2);            // aliases qkv
    unsigned short* gbuf  = (unsigned short*)(ws + OFF_R3);

    // 1) weights -> bf16
    cvt_bf16_kernel<<<(3 * Cc * Cc + 255) / 256, 256, 0, stream>>>(qkv_w,  wqkv,  3 * Cc * Cc);
    cvt_bf16_kernel<<<(Cc * Cc + 255) / 256,     256, 0, stream>>>(proj_w, wproj, Cc * Cc);
    cvt_bf16_kernel<<<(HIDc * Cc + 255) / 256,   256, 0, stream>>>(fc1_w,  wfc1,  HIDc * Cc);
    cvt_bf16_kernel<<<(Cc * HIDc + 255) / 256,   256, 0, stream>>>(fc2_w,  wfc2,  Cc * HIDc);

    // 2) LN1 + shift + window partition
    ln_kernel<<<ROWS, 256, 0, stream>>>(x, n1g, n1b, yw, 1);

    // 3) QKV projection: (ROWS x 512) @ (512 x 1536)
    gemm_bf16_kernel<<<dim3(3 * Cc / 64, ROWS / 128), 256, 0, stream>>>(
        yw, wqkv, qkv_b, ROWS, 3 * Cc, Cc, 0, qkvb, nullptr, nullptr);

    // 4) windowed attention (softmax + bias + mask fused)
    attn_kernel<<<Bc * nWc * HEADS, 128, 0, stream>>>(qkvb, relt, obuf);

    // 5) proj + window reverse + reverse shift + shortcut -> x_res (f32)
    gemm_bf16_kernel<<<dim3(Cc / 64, ROWS / 128), 256, 0, stream>>>(
        obuf, wproj, proj_b, ROWS, Cc, Cc, 3, nullptr, x_res, x);

    // 6) LN2
    ln_kernel<<<ROWS, 256, 0, stream>>>(x_res, n2g, n2b, hbuf, 0);

    // 7) FC1 + GELU
    gemm_bf16_kernel<<<dim3(HIDc / 64, ROWS / 128), 256, 0, stream>>>(
        hbuf, wfc1, fc1_b, ROWS, HIDc, Cc, 1, gbuf, nullptr, nullptr);

    // 8) FC2 + residual -> d_out (f32)
    gemm_bf16_kernel<<<dim3(Cc / 64, ROWS / 128), 256, 0, stream>>>(
        gbuf, wfc2, fc2_b, ROWS, Cc, HIDc, 2, nullptr, (float*)d_out, x_res);
}
